// LocalModel_33440615366872
// MI455X (gfx1250) — matrix operand order repaired and checked by reading, not compiled, not run on hardware
//
#include <hip/hip_runtime.h>

#define NN   10000      // nodes
#define ERAW 160000     // raw edges
#define EDG  170000     // edges + self loops
#define HID  128
#define HH   256        // HEADS*HID

typedef __attribute__((ext_vector_type(16))) __bf16 v16bf;
typedef __attribute__((ext_vector_type(8)))  float  v8f;

__device__ __forceinline__ float lrelu(float x) { return x > 0.f ? x : 0.2f * x; }

__device__ __forceinline__ unsigned short f2bf(float f) {
  unsigned u = __float_as_uint(f);
  u += 0x7fffu + ((u >> 16) & 1u);          // round-to-nearest-even
  return (unsigned short)(u >> 16);
}
// monotonic uint encoding of float for atomicMax-based segment max
__device__ __forceinline__ unsigned enc(float f) {
  unsigned u = __float_as_uint(f);
  return (u & 0x80000000u) ? ~u : (u | 0x80000000u);
}
__device__ __forceinline__ float dec(unsigned u) {
  return __uint_as_float((u & 0x80000000u) ? (u & 0x7fffffffu) : ~u);
}
#define ENC_NEG_INF 0x007FFFFFu

// ---------------------------------------------------------------- prep
__global__ void build_edges(const int* __restrict__ ei, int* __restrict__ s,
                            int* __restrict__ d) {
  int i = blockIdx.x * 256 + threadIdx.x;
  if (i >= EDG) return;
  s[i] = (i < ERAW) ? ei[i]        : (i - ERAW);
  d[i] = (i < ERAW) ? ei[ERAW + i] : (i - ERAW);
}

// Wl/Wr (17,128,256) f32 -> WT (17,256,128) bf16 (transposed for contiguous B frags)
__global__ void convert_wt(const float* __restrict__ Wl, const float* __restrict__ Wr,
                           unsigned short* __restrict__ WTl, unsigned short* __restrict__ WTr) {
  int i = blockIdx.x * 256 + threadIdx.x;
  const int per = 17 * 256 * 128;
  if (i >= 2 * per) return;
  int m = i / per, r = i % per;
  int layer = r / (256 * 128), t = r % (256 * 128);
  int n = t >> 7, k = t & 127;
  const float* W = m ? Wr : Wl;
  unsigned short* O = m ? WTr : WTl;
  O[layer * 32768 + n * 128 + k] = f2bf(W[layer * 32768 + k * 256 + n]);
}

// ---------------------------------------------------------------- dense (WMMA)
// A: activations bf16 [NN,128] row-major. WT: weights bf16 [256,128] (transposed).
// One wave computes FOUR adjacent 16x16 tiles of XL or XR (4-way N blocking:
// one A fragment feeds 4 WMMAs per K-step -> 16 v_wmma per wave).
__global__ __launch_bounds__(256) void dense_wmma(
    const unsigned short* __restrict__ A,
    const unsigned short* __restrict__ WTl, const unsigned short* __restrict__ WTr,
    float* __restrict__ XL, float* __restrict__ XR) {
  int wave = threadIdx.x >> 5, lane = threadIdx.x & 31;
  int mat = wave >> 2;                 // 0:L 1:R
  int ng  = (wave & 3) * 4;            // first of 4 consecutive ntiles
  int mtile = blockIdx.x;
  const unsigned short* WT = mat ? WTr : WTl;
  float* OUT = mat ? XR : XL;

  int arow  = mtile * 16 + (lane & 15);
  int akb   = (lane >> 4) * 8;         // A K-group base (0 or 8)
  int ncol0 = ng * 16 + (lane & 15);
  int bkb   = (lane >> 4) * 8;         // B K-group base (0 or 8)

  union Frag { v16bf v; uint4 q[2]; };
  v8f c0 = {}, c1 = {}, c2 = {}, c3 = {};
  const unsigned short* Ap = A + arow * 128;
#pragma unroll
  for (int k0 = 0; k0 < 128; k0 += 32) {
    Frag a;
    a.q[0] = *(const uint4*)(Ap + k0 + akb);          // K = k0+akb..+7
    a.q[1] = *(const uint4*)(Ap + k0 + 16 + akb);     // K = k0+16+akb..+7
    const unsigned short* Bp = WT + ncol0 * 128 + k0 + bkb;
    Frag b0, b1, b2, b3;
    b0.q[0] = *(const uint4*)(Bp);
    b0.q[1] = *(const uint4*)(Bp + 16);
    b1.q[0] = *(const uint4*)(Bp + 16 * 128);
    b1.q[1] = *(const uint4*)(Bp + 16 * 128 + 16);
    b2.q[0] = *(const uint4*)(Bp + 32 * 128);
    b2.q[1] = *(const uint4*)(Bp + 32 * 128 + 16);
    b3.q[0] = *(const uint4*)(Bp + 48 * 128);
    b3.q[1] = *(const uint4*)(Bp + 48 * 128 + 16);
    c0 = __builtin_amdgcn_wmma_f32_16x16x32_bf16(false, a.v, false, b0.v,
                                                 (short)0, c0, false, false);
    c1 = __builtin_amdgcn_wmma_f32_16x16x32_bf16(false, a.v, false, b1.v,
                                                 (short)0, c1, false, false);
    c2 = __builtin_amdgcn_wmma_f32_16x16x32_bf16(false, a.v, false, b2.v,
                                                 (short)0, c2, false, false);
    c3 = __builtin_amdgcn_wmma_f32_16x16x32_bf16(false, a.v, false, b3.v,
                                                 (short)0, c3, false, false);
  }
  int rbase = mtile * 16 + ((lane >> 4) << 3); // D: VGPR r -> M = r + 8*(lane>=16)
  float* O = OUT + rbase * 256 + ncol0;
#pragma unroll
  for (int r = 0; r < 8; ++r) {
    O[r * 256 +  0] = c0[r];
    O[r * 256 + 16] = c1[r];
    O[r * 256 + 32] = c2[r];
    O[r * 256 + 48] = c3[r];
  }
}

// first layer: x [NN,3] f32 -> XL/XR [NN,256]
__global__ void dense_first(const float* __restrict__ x, const float* __restrict__ Wl,
                            const float* __restrict__ Wr, float* __restrict__ XL,
                            float* __restrict__ XR) {
  int i = blockIdx.x * 256 + threadIdx.x;
  if (i >= NN * 512) return;
  int n = i >> 9, j = i & 511, col = j & 255;
  const float* W = (j & 256) ? Wr : Wl;
  const float* xp = x + n * 3;
  float v = xp[0] * W[col] + xp[1] * W[256 + col] + xp[2] * W[512 + col];
  ((j & 256) ? XR : XL)[n * 256 + col] = v;
}

// last layer: hin [NN,128] f32 -> xlL/xrL [NN,2]
__global__ void dense_last(const float* __restrict__ hin, const float* __restrict__ Wl,
                           const float* __restrict__ Wr, float* __restrict__ xlL,
                           float* __restrict__ xrL) {
  int i = blockIdx.x * 256 + threadIdx.x;
  if (i >= NN * 4) return;
  int n = i >> 2, j = i & 3, h = j & 1;
  const float* W = (j >> 1) ? Wr : Wl;
  float s = 0.f;
#pragma unroll 8
  for (int k = 0; k < 128; ++k) s += hin[n * 128 + k] * W[k * 2 + h];
  ((j >> 1) ? xrL : xlL)[n * 2 + h] = s;
}

// ---------------------------------------------------------------- edge phase
__global__ void init_mden(unsigned* __restrict__ mkey, float* __restrict__ den, int n2) {
  int i = blockIdx.x * 256 + threadIdx.x;
  if (i >= n2) return;
  mkey[i] = ENC_NEG_INF;
  den[i] = 0.f;
}

// wave per (edge, head); lane handles 4 channels
__global__ __launch_bounds__(256) void edge_logits(
    const int* __restrict__ src, const int* __restrict__ dst,
    const float* __restrict__ XL, const float* __restrict__ XR,
    const float* __restrict__ att, float* __restrict__ logits,
    unsigned* __restrict__ mkey) {
  int wave = threadIdx.x >> 5, lane = threadIdx.x & 31;
  int e = blockIdx.x * 4 + (wave >> 1);
  if (e >= EDG) return;
  int head = wave & 1;
  int s = src[e], d = dst[e];
  int cb = head * 128 + lane * 4;
  float4 a = *(const float4*)(XL + s * 256 + cb);
  float4 b = *(const float4*)(XR + d * 256 + cb);
  float4 w = *(const float4*)(att + cb);
  float acc = lrelu(a.x + b.x) * w.x + lrelu(a.y + b.y) * w.y +
              lrelu(a.z + b.z) * w.z + lrelu(a.w + b.w) * w.w;
  for (int off = 16; off; off >>= 1) acc += __shfl_xor(acc, off, 32);
  if (lane == 0) {
    logits[e * 2 + head] = acc;
    atomicMax(mkey + d * 2 + head, enc(acc));
  }
}

__global__ void expden(const int* __restrict__ dst, const unsigned* __restrict__ mkey,
                       const float* __restrict__ logits, float* __restrict__ exv,
                       float* __restrict__ den) {
  int i = blockIdx.x * 256 + threadIdx.x;
  if (i >= EDG * 2) return;
  int e = i >> 1, h = i & 1, d = dst[e];
  float ex = __expf(logits[i] - dec(mkey[d * 2 + h]));
  exv[i] = ex;
  atomicAdd(den + d * 2 + h, ex);
}

__global__ __launch_bounds__(256) void edge_agg(
    const int* __restrict__ src, const int* __restrict__ dst,
    const float* __restrict__ XL, const float* __restrict__ exv,
    const float* __restrict__ den, float* __restrict__ acc) {
  int wave = threadIdx.x >> 5, lane = threadIdx.x & 31;
  int e = blockIdx.x * 4 + (wave >> 1);
  if (e >= EDG) return;
  int head = wave & 1;
  int s = src[e], d = dst[e];
  int cb = head * 128 + lane * 4;
  float alpha = exv[e * 2 + head] / (den[d * 2 + head] + 1e-16f);
  float4 a = *(const float4*)(XL + s * 256 + cb);
  float* o = acc + d * 256 + cb;
  atomicAdd(o + 0, alpha * a.x);
  atomicAdd(o + 1, alpha * a.y);
  atomicAdd(o + 2, alpha * a.z);
  atomicAdd(o + 3, alpha * a.w);
}

// ---------------------------------------------------------------- BN + act
__global__ void bnstats(const float* __restrict__ acc, float* __restrict__ hcur,
                        float* __restrict__ sum, float* __restrict__ sq) {
  int c = threadIdx.x;  // 128
  float s = 0.f, q = 0.f;
  for (int n = blockIdx.x; n < NN; n += gridDim.x) {
    float h = 0.5f * (acc[n * 256 + c] + acc[n * 256 + 128 + c]);  // mean over heads
    hcur[n * 128 + c] = h;
    s += h; q += h * h;
  }
  atomicAdd(sum + c, s);
  atomicAdd(sq + c, q);
}

__global__ void bnnorm(float* __restrict__ hcur, unsigned short* __restrict__ hb16,
                       const float* __restrict__ sum, const float* __restrict__ sq,
                       const float* __restrict__ g, const float* __restrict__ b,
                       float* __restrict__ skip) {
  int i = blockIdx.x * 256 + threadIdx.x;
  if (i >= NN * 128) return;
  int c = i & 127;
  float mu = sum[c] * (1.f / NN);
  float var = sq[c] * (1.f / NN) - mu * mu;
  float y = lrelu((hcur[i] - mu) * rsqrtf(var + 1e-5f) * g[c] + b[c]);
  hcur[i] = y;
  hb16[i] = f2bf(y);
  if (skip) skip[i] = y;
}

__global__ void addskip(const float* __restrict__ hcur, const float* __restrict__ xs,
                        float* __restrict__ hin, unsigned short* __restrict__ hb16) {
  int i = blockIdx.x * 256 + threadIdx.x;
  if (i >= NN * 128) return;
  float v = hcur[i] + xs[i];
  hin[i] = v;
  hb16[i] = f2bf(v);
}

// ---------------------------------------------------------------- last block (C=1)
__global__ void last_logits(const int* __restrict__ src, const int* __restrict__ dst,
                            const float* __restrict__ xl, const float* __restrict__ xr,
                            const float* __restrict__ attL, float* __restrict__ logits,
                            unsigned* __restrict__ mkey) {
  int i = blockIdx.x * 256 + threadIdx.x;
  if (i >= EDG * 2) return;
  int e = i >> 1, h = i & 1;
  int s = src[e], d = dst[e];
  float v = lrelu(xl[s * 2 + h] + xr[d * 2 + h]) * attL[h];
  logits[i] = v;
  atomicMax(mkey + d * 2 + h, enc(v));
}

__global__ void last_agg(const int* __restrict__ src, const int* __restrict__ dst,
                         const float* __restrict__ xl, const float* __restrict__ exv,
                         const float* __restrict__ den, float* __restrict__ accL) {
  int i = blockIdx.x * 256 + threadIdx.x;
  if (i >= EDG * 2) return;
  int e = i >> 1, h = i & 1;
  int s = src[e], d = dst[e];
  float alpha = exv[i] / (den[d * 2 + h] + 1e-16f);
  atomicAdd(accL + d * 2 + h, alpha * xl[s * 2 + h]);
}

__global__ void last_stats(const float* __restrict__ accL, float* __restrict__ hL,
                           float* __restrict__ sumsq) {
  int tid = blockIdx.x * 128 + threadIdx.x;
  float s = 0.f, q = 0.f;
  for (int n = tid; n < NN; n += 64 * 128) {
    float h = 0.5f * (accL[n * 2] + accL[n * 2 + 1]);
    hL[n] = h;
    s += h; q += h * h;
  }
  for (int off = 16; off; off >>= 1) {
    s += __shfl_xor(s, off, 32);
    q += __shfl_xor(q, off, 32);
  }
  if ((threadIdx.x & 31) == 0) {
    atomicAdd(sumsq + 0, s);
    atomicAdd(sumsq + 1, q);
  }
}

__global__ void last_out(const float* __restrict__ hL, const float* __restrict__ sumsq,
                         const float* __restrict__ gL, const float* __restrict__ bL,
                         float* __restrict__ out) {
  int n = blockIdx.x * 256 + threadIdx.x;
  if (n >= NN) return;
  float mu = sumsq[0] * (1.f / NN);
  float var = sumsq[1] * (1.f / NN) - mu * mu;
  out[n] = lrelu((hL[n] - mu) * rsqrtf(var + 1e-5f) * gL[0] + bL[0]);
}

// ---------------------------------------------------------------- host
extern "C" void kernel_launch(void* const* d_in, const int* in_sizes, int n_in,
                              void* d_out, int out_size, void* d_ws, size_t ws_size,
                              hipStream_t stream) {
  const float* x    = (const float*)d_in[0];
  const int*   ei   = (const int*)  d_in[1];
  const float* Wl1  = (const float*)d_in[2];
  const float* Wr1  = (const float*)d_in[3];
  const float* att1 = (const float*)d_in[4];
  const float* g1   = (const float*)d_in[5];
  const float* b1   = (const float*)d_in[6];
  const float* Wlm  = (const float*)d_in[7];
  const float* Wrm  = (const float*)d_in[8];
  const float* attm = (const float*)d_in[9];
  const float* gm   = (const float*)d_in[10];
  const float* bm   = (const float*)d_in[11];
  const float* WlL  = (const float*)d_in[12];
  const float* WrL  = (const float*)d_in[13];
  const float* attL = (const float*)d_in[14];
  const float* gL   = (const float*)d_in[15];
  const float* bL   = (const float*)d_in[16];
  float* out = (float*)d_out;

  char* p = (char*)d_ws;
  auto alloc = [&](size_t bytes) -> char* {
    char* r = p;
    p += (bytes + 255) & ~(size_t)255;
    return r;
  };
  int*            srcsl = (int*)alloc((size_t)EDG * 4);
  int*            dstsl = (int*)alloc((size_t)EDG * 4);
  unsigned short* WTl   = (unsigned short*)alloc((size_t)17 * 256 * 128 * 2);
  unsigned short* WTr   = (unsigned short*)alloc((size_t)17 * 256 * 128 * 2);
  unsigned short* hb16  = (unsigned short*)alloc((size_t)NN * 128 * 2);
  float*          hcur  = (float*)alloc((size_t)NN * 128 * 4);
  float*          hin   = (float*)alloc((size_t)NN * 128 * 4);
  float*          XL    = (float*)alloc((size_t)NN * 256 * 4);
  float*          XR    = (float*)alloc((size_t)NN * 256 * 4);
  float*          logits= (float*)alloc((size_t)EDG * 2 * 4);
  float*          exv   = (float*)alloc((size_t)EDG * 2 * 4);
  unsigned*       mkey  = (unsigned*)alloc((size_t)NN * 2 * 4);
  float*          den   = (float*)alloc((size_t)NN * 2 * 4);
  float*          acc   = (float*)alloc((size_t)NN * 256 * 4);
  float*          sumsq = (float*)alloc(256 * 4);   // sum[128] | sq[128]
  float*          xs    = (float*)alloc((size_t)9 * NN * 128 * 4);
  float*          xlL   = (float*)alloc((size_t)NN * 2 * 4);
  float*          xrL   = (float*)alloc((size_t)NN * 2 * 4);
  float*          accL  = (float*)alloc((size_t)NN * 2 * 4);

  // ---- prep
  build_edges<<<(EDG + 255) / 256, 256, 0, stream>>>(ei, srcsl, dstsl);
  convert_wt<<<(2 * 17 * 256 * 128 + 255) / 256, 256, 0, stream>>>(Wlm, Wrm, WTl, WTr);

  auto run_edges = [&](const float* attp) {
    init_mden<<<(NN * 2 + 255) / 256, 256, 0, stream>>>(mkey, den, NN * 2);
    hipMemsetAsync(acc, 0, (size_t)NN * 256 * 4, stream);
    edge_logits<<<EDG / 4, 256, 0, stream>>>(srcsl, dstsl, XL, XR, attp, logits, mkey);
    expden<<<(EDG * 2 + 255) / 256, 256, 0, stream>>>(dstsl, mkey, logits, exv, den);
    edge_agg<<<EDG / 4, 256, 0, stream>>>(srcsl, dstsl, XL, exv, den, acc);
  };
  auto run_post = [&](const float* gp, const float* bp, float* skip) {
    hipMemsetAsync(sumsq, 0, 256 * 4, stream);
    bnstats<<<64, 128, 0, stream>>>(acc, hcur, sumsq, sumsq + 128);
    bnnorm<<<(NN * 128 + 255) / 256, 256, 0, stream>>>(hcur, hb16, sumsq, sumsq + 128,
                                                       gp, bp, skip);
  };

  // ---- block 0 (input, K=3)
  dense_first<<<(NN * 512 + 255) / 256, 256, 0, stream>>>(x, Wl1, Wr1, XL, XR);
  run_edges(att1);
  run_post(g1, b1, xs);  // xs[0]

  // ---- down path: mid layers 0..8
  for (int i = 0; i < 9; ++i) {
    dense_wmma<<<NN / 16, 256, 0, stream>>>(hb16, WTl + (size_t)i * 32768,
                                            WTr + (size_t)i * 32768, XL, XR);
    run_edges(attm + i * 256);
    run_post(gm + i * 128, bm + i * 128,
             (i < 8) ? xs + (size_t)(i + 1) * NN * 128 : nullptr);
  }

  // ---- up path: mid layers 9..16 with U-net skip adds
  for (int j = 0; j < 8; ++j) {
    addskip<<<(NN * 128 + 255) / 256, 256, 0, stream>>>(
        hcur, xs + (size_t)(8 - j) * NN * 128, hin, hb16);
    int k = 9 + j;
    dense_wmma<<<NN / 16, 256, 0, stream>>>(hb16, WTl + (size_t)k * 32768,
                                            WTr + (size_t)k * 32768, XL, XR);
    run_edges(attm + k * 256);
    run_post(gm + k * 128, bm + k * 128, nullptr);
  }

  // ---- final block (C=1)
  addskip<<<(NN * 128 + 255) / 256, 256, 0, stream>>>(hcur, xs, hin, hb16);
  dense_last<<<(NN * 4 + 255) / 256, 256, 0, stream>>>(hin, WlL, WrL, xlL, xrL);
  init_mden<<<(NN * 2 + 255) / 256, 256, 0, stream>>>(mkey, den, NN * 2);
  hipMemsetAsync(accL, 0, (size_t)NN * 2 * 4, stream);
  last_logits<<<(EDG * 2 + 255) / 256, 256, 0, stream>>>(srcsl, dstsl, xlL, xrL, attL,
                                                         logits, mkey);
  expden<<<(EDG * 2 + 255) / 256, 256, 0, stream>>>(dstsl, mkey, logits, exv, den);
  last_agg<<<(EDG * 2 + 255) / 256, 256, 0, stream>>>(srcsl, dstsl, xlL, exv, den, accL);
  hipMemsetAsync(sumsq, 0, 8, stream);
  last_stats<<<64, 128, 0, stream>>>(accL, hin, sumsq);
  last_out<<<(NN + 255) / 256, 256, 0, stream>>>(hin, sumsq, gL, bL, out);
}
